// DivergenceRN_1322849927392
// MI455X (gfx1250) — compile-verified, matrix operand order checked by reading, not run on hardware
//
#include <hip/hip_runtime.h>
#include <math.h>

#define B_ 4
#define N_ 384
#define D_ 64
#define H_ 64

typedef __attribute__((ext_vector_type(16))) _Float16 v16h;
typedef __attribute__((ext_vector_type(8)))  _Float16 v8h;
typedef __attribute__((ext_vector_type(8)))  float    v8f;

// ---------------------------------------------------------------------------
// Phase 1: pa' = A @ W1[:D] + b1  and  pc = C @ W1[D:]  for both live branches,
// stored as f16 (the WMMA input precision) so phase 2 can use packed-f16 math.
// grid = (N, B, 4 matrices), block = 64 threads (one per output feature h).
// mat 0: paXX (X, W_xx1[:D], +b_xx1)   mat 1: pcXX (X, W_xx1[D:])
// mat 2: paYX (Y, W_yx1[:D], +b_yx1)   mat 3: pcYX (X, W_yx1[D:])
// ---------------------------------------------------------------------------
__global__ void pair_lin1_kernel(const float* __restrict__ X,
                                 const float* __restrict__ Y,
                                 const float* __restrict__ Wxx1, const float* __restrict__ bxx1,
                                 const float* __restrict__ Wyx1, const float* __restrict__ byx1,
                                 _Float16* __restrict__ paXX, _Float16* __restrict__ pcXX,
                                 _Float16* __restrict__ paYX, _Float16* __restrict__ pcYX)
{
    const int j   = blockIdx.x;
    const int b   = blockIdx.y;
    const int mat = blockIdx.z;
    const int h   = threadIdx.x;

    const float* in; const float* W; const float* bias; _Float16* out;
    switch (mat) {
      case 0:  in = X; W = Wxx1;           bias = bxx1;    out = paXX; break;
      case 1:  in = X; W = Wxx1 + D_ * H_; bias = nullptr; out = pcXX; break;
      case 2:  in = Y; W = Wyx1;           bias = byx1;    out = paYX; break;
      default: in = X; W = Wyx1 + D_ * H_; bias = nullptr; out = pcYX; break;
    }

    __shared__ float srow[D_];
    srow[h] = in[(b * N_ + j) * D_ + h];
    __syncthreads();

    float acc = bias ? bias[h] : 0.0f;
    #pragma unroll
    for (int d = 0; d < D_; ++d)
        acc = fmaf(srow[d], W[d * H_ + h], acc);
    out[(b * N_ + j) * H_ + h] = (_Float16)acc;
}

// ---------------------------------------------------------------------------
// Phase 2 (hot loop): per (b, i, branch), compute
//   Z[b,i,:] = max_j ( relu(pa'[b,j,:] + pc[b,i,:]) @ W2 ) + b2
// via v_wmma_f32_16x16x32_f16, 4 waves/block, 6 of the 24 j-tiles per wave.
// A fragments are built with packed-f16 add/max from two contiguous 16-byte
// loads per fragment; W2 lives as 8 resident B fragments; the 16x16 f32 D
// tiles are max-reduced immediately (no [B,N,N,H] tensor materialized).
// ---------------------------------------------------------------------------
__global__ __launch_bounds__(128) void pair_wmma_max_kernel(
    const _Float16* __restrict__ paXX, const _Float16* __restrict__ pcXX,
    const _Float16* __restrict__ paYX, const _Float16* __restrict__ pcYX,
    const float* __restrict__ Wxx2, const float* __restrict__ bxx2,
    const float* __restrict__ Wyx2, const float* __restrict__ byx2,
    float* __restrict__ Zxx, float* __restrict__ Zyx)
{
    const int i  = blockIdx.x;
    const int b  = blockIdx.y;
    const int br = blockIdx.z;

    const _Float16* pa = br ? paYX : paXX;
    const _Float16* pc = br ? pcYX : pcXX;
    const float*    W2 = br ? Wyx2 : Wxx2;
    const float*    b2 = br ? byx2 : bxx2;
    float*          Z  = br ? Zyx  : Zxx;

    const int tid  = threadIdx.x;
    const int wave = tid >> 5;
    const int lane = tid & 31;
    const int mrow = lane & 15;          // row (A) / column (B) within tile
    const int kgrp = (lane >> 4) << 3;   // K sub-group: 0 or 8

    // --- Resident B fragments: W2 as 2 (K-halves) x 4 (N-tiles) of 32x16 f16.
    // Lane holds column N = nt*16 + mrow; K pattern mirrors the A layout.
    v16h bf[2][4];
    #pragma unroll
    for (int kh = 0; kh < 2; ++kh) {
      #pragma unroll
      for (int nt = 0; nt < 4; ++nt) {
        const int n = nt * 16 + mrow;
        #pragma unroll
        for (int v = 0; v < 8; ++v) {
          const int k0 = kh * 32 + ((v & 4) << 2) + kgrp + ((v & 3) << 1);
          bf[kh][nt][2 * v]     = (_Float16)W2[(k0    ) * H_ + n];
          bf[kh][nt][2 * v + 1] = (_Float16)W2[(k0 + 1) * H_ + n];
        }
      }
    }

    // --- Preload this lane's pc[b,i,k] as packed f16: per K-half, the lane's
    // 16 K-values are two contiguous runs of 8 (kgrp..kgrp+7, kgrp+16..+23).
    const _Float16* pcRow = pc + (size_t)(b * N_ + i) * H_;
    v8h pcv[2][2];
    #pragma unroll
    for (int kh = 0; kh < 2; ++kh) {
      pcv[kh][0] = *(const v8h*)(pcRow + kh * 32 + kgrp);
      pcv[kh][1] = *(const v8h*)(pcRow + kh * 32 + kgrp + 16);
    }
    const v8h zero8 = (v8h)(_Float16)0.0f;

    float runmax[4] = {-3.402823466e+38f, -3.402823466e+38f,
                       -3.402823466e+38f, -3.402823466e+38f};

    // --- 6 j-tiles per wave (24 total).
    const int jt0 = wave * 6;
    for (int jt = jt0; jt < jt0 + 6; ++jt) {
      const _Float16* paRow = pa + (size_t)(b * N_ + jt * 16 + mrow) * H_;
      v16h a[2];
      #pragma unroll
      for (int kh = 0; kh < 2; ++kh) {
        v8h lo = *(const v8h*)(paRow + kh * 32 + kgrp);        // 16B load
        v8h hi = *(const v8h*)(paRow + kh * 32 + kgrp + 16);   // 16B load
        lo = __builtin_elementwise_max(lo + pcv[kh][0], zero8); // v_pk_add/max
        hi = __builtin_elementwise_max(hi + pcv[kh][1], zero8);
        a[kh] = __builtin_shufflevector(lo, hi, 0, 1, 2, 3, 4, 5, 6, 7,
                                        8, 9, 10, 11, 12, 13, 14, 15);
      }
      #pragma unroll
      for (int nt = 0; nt < 4; ++nt) {
        v8f c = {};
        c = __builtin_amdgcn_wmma_f32_16x16x32_f16(false, a[0], false, bf[0][nt],
                                                   (short)0, c, false, false);
        c = __builtin_amdgcn_wmma_f32_16x16x32_f16(false, a[1], false, bf[1][nt],
                                                   (short)0, c, false, false);
        // Immediate max over the 8 M-rows (j values) this lane holds.
        #pragma unroll
        for (int r = 0; r < 8; ++r)
          runmax[nt] = fmaxf(runmax[nt], c[r]);
      }
    }

    // --- Combine the two half-wave M-halves, then the 4 waves via LDS.
    __shared__ float smax[4][H_];
    #pragma unroll
    for (int nt = 0; nt < 4; ++nt) {
      const float other = __shfl_xor(runmax[nt], 16, 32);
      const float full  = fmaxf(runmax[nt], other);
      if (lane < 16)
        smax[wave][nt * 16 + mrow] = full;
    }
    __syncthreads();

    if (tid < H_) {
      const float m01 = fmaxf(smax[0][tid], smax[1][tid]);
      const float m23 = fmaxf(smax[2][tid], smax[3][tid]);
      Z[(size_t)(b * N_ + i) * H_ + tid] = fmaxf(m01, m23) + b2[tid];
    }
}

// ---------------------------------------------------------------------------
// Phase 3: S[b,h] = sum_i (Zxx - Zyx);  z = [S, S];  out = relu(z@Wd1+bd1)@Wd2+bd2
// grid = B, block = 64.
// ---------------------------------------------------------------------------
__global__ void decoder_kernel(const float* __restrict__ Zxx, const float* __restrict__ Zyx,
                               const float* __restrict__ Wd1, const float* __restrict__ bd1,
                               const float* __restrict__ Wd2, const float* __restrict__ bd2,
                               float* __restrict__ out)
{
    const int b = blockIdx.x;
    const int h = threadIdx.x;

    __shared__ float sS[H_];
    __shared__ float sH[H_];

    float s = 0.0f;
    for (int i = 0; i < N_; ++i) {
      const size_t idx = (size_t)(b * N_ + i) * H_ + h;
      s += Zxx[idx] - Zyx[idx];
    }
    sS[h] = s;
    __syncthreads();

    float acc = bd1[h];
    #pragma unroll 4
    for (int q = 0; q < 2 * H_; ++q)
      acc = fmaf(sS[q & (H_ - 1)], Wd1[q * H_ + h], acc);
    sH[h] = fmaxf(acc, 0.0f) * Wd2[h];
    __syncthreads();

    if (h == 0) {
      float t = bd2[0];
      #pragma unroll
      for (int q = 0; q < H_; ++q) t += sH[q];
      out[b] = t;
    }
}

// ---------------------------------------------------------------------------
extern "C" void kernel_launch(void* const* d_in, const int* in_sizes, int n_in,
                              void* d_out, int out_size, void* d_ws, size_t ws_size,
                              hipStream_t stream) {
    (void)in_sizes; (void)n_in; (void)out_size; (void)ws_size;
    const float* X    = (const float*)d_in[0];
    const float* Y    = (const float*)d_in[1];
    const float* Wxx1 = (const float*)d_in[2];
    const float* bxx1 = (const float*)d_in[3];
    const float* Wxx2 = (const float*)d_in[4];
    const float* bxx2 = (const float*)d_in[5];
    // d_in[6..9]  : xy branch (dead code in the reference)
    const float* Wyx1 = (const float*)d_in[10];
    const float* byx1 = (const float*)d_in[11];
    const float* Wyx2 = (const float*)d_in[12];
    const float* byx2 = (const float*)d_in[13];
    // d_in[14..17]: yy branch (dead code in the reference)
    const float* Wd1  = (const float*)d_in[18];
    const float* bd1  = (const float*)d_in[19];
    const float* Wd2  = (const float*)d_in[20];
    const float* bd2  = (const float*)d_in[21];

    const size_t S = (size_t)B_ * N_ * H_;   // 98304 elements per buffer
    _Float16* hws = (_Float16*)d_ws;
    _Float16* paXX = hws;
    _Float16* pcXX = hws + 1 * S;
    _Float16* paYX = hws + 2 * S;
    _Float16* pcYX = hws + 3 * S;
    float* fws = (float*)(hws + 4 * S);
    float* Zxx = fws;
    float* Zyx = fws + S;

    pair_lin1_kernel<<<dim3(N_, B_, 4), 64, 0, stream>>>(
        X, Y, Wxx1, bxx1, Wyx1, byx1, paXX, pcXX, paYX, pcYX);

    pair_wmma_max_kernel<<<dim3(N_, B_, 2), 128, 0, stream>>>(
        paXX, pcXX, paYX, pcYX, Wxx2, bxx2, Wyx2, byx2, Zxx, Zyx);

    decoder_kernel<<<B_, H_, 0, stream>>>(
        Zxx, Zyx, Wd1, bd1, Wd2, bd2, (float*)d_out);
}
